// MTA_66812511257225
// MI455X (gfx1250) — matrix operand order repaired and checked by reading, not compiled, not run on hardware
//
#include <hip/hip_runtime.h>
#include <hip/hip_bf16.h>
#include <math.h>

#define B_ 32
#define T_ 2000
#define E_ 1024
#define A_ 1024
#define TINY_ 1.1754943508222875e-38f

typedef __attribute__((ext_vector_type(16))) __bf16 v16bf;
typedef __attribute__((ext_vector_type(8)))  float  v8f;

union V16BF { v16bf v; uint4 q[2]; };

// ---- helpers -------------------------------------------------------------
__device__ __forceinline__ unsigned short f2bf_rne(float f) {
  unsigned int u = __float_as_uint(f);
  u += 0x7FFFu + ((u >> 16) & 1u);
  return (unsigned short)(u >> 16);
}

// pack two f32 -> two bf16 (truncation) in one v_perm_b32
__device__ __forceinline__ unsigned int pack_bf16_tr(float f0, float f1) {
#if __has_builtin(__builtin_amdgcn_perm)
  return __builtin_amdgcn_perm(__float_as_uint(f1), __float_as_uint(f0), 0x07060302u);
#else
  return (__float_as_uint(f0) >> 16) | (__float_as_uint(f1) & 0xFFFF0000u);
#endif
}

// ---- 1) W_enc f32 -> bf16 (one time, RNE) --------------------------------
__global__ void __launch_bounds__(256) k_convert_w(
    const float* __restrict__ W, unsigned short* __restrict__ Wb) {
  int i = blockIdx.x * 256 + threadIdx.x;           // float4 index, 262144 total
  float4 x = ((const float4*)W)[i];
  ushort4 o;
  o.x = f2bf_rne(x.x); o.y = f2bf_rne(x.y);
  o.z = f2bf_rne(x.z); o.w = f2bf_rne(x.w);
  ((ushort4*)Wb)[i] = o;
}

// ---- 2) scale = mono_factor / ||g|| --------------------------------------
__global__ void __launch_bounds__(256) k_scale(
    const float* __restrict__ g, const float* __restrict__ mono_factor,
    float* __restrict__ out) {
  __shared__ float red[256];
  int tid = threadIdx.x;
  float s = 0.f;
  for (int i = tid; i < A_; i += 256) { float v = g[i]; s += v * v; }
  red[tid] = s; __syncthreads();
  for (int off = 128; off > 0; off >>= 1) {
    if (tid < off) red[tid] += red[tid + off];
    __syncthreads();
  }
  if (tid == 0) out[0] = mono_factor[0] / sqrtf(red[0]);
}

// ---- 3) r[b,a] = dec_z[b,:] . W_dec[a,:] + b_enc[a] ----------------------
__global__ void __launch_bounds__(256) k_decproj(
    const float* __restrict__ dec_z, const float* __restrict__ W_dec,
    const float* __restrict__ b_enc, float* __restrict__ r) {
  int a = blockIdx.x * 8 + (threadIdx.x >> 5);      // 128 blocks -> a in [0,1024)
  int b = threadIdx.x & 31;
  const float* wrow = W_dec + (size_t)a * E_;       // broadcast across the wave
  const float* drow = dec_z + (size_t)b * E_;
  float acc = 0.f;
  for (int e = 0; e < E_; e += 4) {
    float4 wv = *(const float4*)(wrow + e);
    float4 dv = *(const float4*)(drow + e);
    acc += wv.x * dv.x + wv.y * dv.y + wv.z * dv.z + wv.w * dv.w;
  }
  r[(size_t)b * A_ + a] = acc + b_enc[a];
}

// ---- 4) fused GEMM + tanh-reduce energy kernel (WMMA bf16) ---------------
// grid 500 blocks x 256 threads; wave w handles M-tile (16 rows) mtile=bid*8+w
__global__ void __launch_bounds__(256, 1) k_energy(
    const float* __restrict__ enc, const unsigned short* __restrict__ Wb,
    const float* __restrict__ rba, const float* __restrict__ g,
    const float* __restrict__ scale_p, const float* __restrict__ mono_bias,
    const int* __restrict__ len, const float* __restrict__ noise,
    float* __restrict__ e_buf) {
  const int tid  = threadIdx.x;
  const int w    = tid >> 5;
  const int lane = tid & 31;
  const int half = lane >> 4;       // 0: K 0-7/16-23 (A), K 0-7/16-23 (B)
  const int l16  = lane & 15;
  const int gr0  = (blockIdx.x * 8 + w) * 16;       // first global row of tile
  const int b    = gr0 / T_;                        // T_ % 16 == 0: tile in one b

  const float* arow = enc + (size_t)(gr0 + l16) * E_;  // lane's A row (M=l16)

  float es[8];
#pragma unroll
  for (int j = 0; j < 8; ++j) es[j] = 0.f;

  const float scale = scale_p[0];
  const float bias  = mono_bias[0];
  const v8f vzero = {0.f, 0.f, 0.f, 0.f, 0.f, 0.f, 0.f, 0.f};

  for (int pass = 0; pass < 4; ++pass) {            // 4 x 16 N-tiles = 1024 cols
    v8f acc[16];
#pragma unroll
    for (int n = 0; n < 16; ++n) acc[n] = vzero;

    for (int kc = 0; kc < 32; ++kc) {               // K chunks of 32
      const int k0 = kc * 32;
      // ---- A fragment: row l16, halves = [kb..kb+7] then [kb+16..kb+23]
      const int kb = k0 + half * 8;
      float4 x0 = *(const float4*)(arow + kb);
      float4 x1 = *(const float4*)(arow + kb + 4);
      float4 y0 = *(const float4*)(arow + kb + 16);
      float4 y1 = *(const float4*)(arow + kb + 20);
      V16BF af;
      af.q[0].x = pack_bf16_tr(x0.x, x0.y);
      af.q[0].y = pack_bf16_tr(x0.z, x0.w);
      af.q[0].z = pack_bf16_tr(x1.x, x1.y);
      af.q[0].w = pack_bf16_tr(x1.z, x1.w);
      af.q[1].x = pack_bf16_tr(y0.x, y0.y);
      af.q[1].y = pack_bf16_tr(y0.z, y0.w);
      af.q[1].z = pack_bf16_tr(y1.x, y1.y);
      af.q[1].w = pack_bf16_tr(y1.z, y1.w);

      const int eoff = k0 + half * 8;               // B: halves = [kb..kb+7] then [kb+16..kb+23]
#pragma unroll
      for (int n = 0; n < 16; ++n) {
        const int a = pass * 256 + n * 16 + l16;    // N column = W_enc row
        const unsigned short* bp = Wb + (size_t)a * E_ + eoff;
        V16BF bf;
        bf.q[0] = *(const uint4*)(bp);
        bf.q[1] = *(const uint4*)(bp + 16);
        acc[n] = __builtin_amdgcn_wmma_f32_16x16x32_bf16(
            false, af.v, false, bf.v, (short)0, acc[n], false, false);
      }
    }

    // epilogue: fold this pass's 256 a-columns through tanh * g into es[]
#pragma unroll
    for (int n = 0; n < 16; ++n) {
      const int a   = pass * 256 + n * 16 + l16;
      const float ga = g[a];
      const float rv = rba[(size_t)b * A_ + a];
#pragma unroll
      for (int j = 0; j < 8; ++j)
        es[j] += tanhf(acc[n][j] + rv) * ga;        // row = j + 8*half
    }
    __syncthreads();                                // keep waves lockstep for WGP$ reuse
  }

  // reduce over the 16 lanes of each half (wave32 butterfly)
#pragma unroll
  for (int j = 0; j < 8; ++j) {
    float v = es[j];
    v += __shfl_xor(v, 1, 32);
    v += __shfl_xor(v, 2, 32);
    v += __shfl_xor(v, 4, 32);
    v += __shfl_xor(v, 8, 32);
    es[j] = v;
  }

  if (l16 == 0) {                                   // lanes 0 (rows 0-7), 16 (rows 8-15)
    const int tlen = len[b];
#pragma unroll
    for (int j = 0; j < 8; ++j) {
      int gr = gr0 + j + half * 8;
      int t  = gr - b * T_;
      float e = (t >= tlen) ? -__builtin_inff()
                            : scale * es[j] + bias + noise[(size_t)b * T_ + t];
      e_buf[gr] = e;
    }
  }
}

// ---- 5) per-row sigmoid + exclusive log-cumprod scan ---------------------
__global__ void __launch_bounds__(256) k_scan(
    const float* __restrict__ e_buf, float* __restrict__ a_out) {
  const int b = blockIdx.x, tid = threadIdx.x;
  __shared__ float ssum[256];
  float p[8], l[8];
#pragma unroll
  for (int j = 0; j < 8; ++j) {
    int t = tid * 8 + j;
    if (t < T_) {
      float e  = e_buf[b * T_ + t];
      float pp = 1.f / (1.f + expf(-e));            // e=-inf -> p=0
      p[j] = pp;
      float omp = fminf(fmaxf(1.f - pp, TINY_), 1.f);
      l[j] = logf(omp);
    } else { p[j] = 0.f; l[j] = 0.f; }
  }
  float s = 0.f;
#pragma unroll
  for (int j = 0; j < 8; ++j) s += l[j];
  ssum[tid] = s; __syncthreads();
  for (int off = 1; off < 256; off <<= 1) {         // Hillis-Steele inclusive scan
    float v = (tid >= off) ? ssum[tid - off] : 0.f;
    __syncthreads();
    ssum[tid] += v;
    __syncthreads();
  }
  float c = (tid == 0) ? 0.f : ssum[tid - 1];       // exclusive prefix
#pragma unroll
  for (int j = 0; j < 8; ++j) {
    int t = tid * 8 + j;
    if (t < T_) a_out[(size_t)b * T_ + t] = p[j] * expf(c);
    c += l[j];
  }
}

// ---- 6) context partials: cpart[b,ch,:] = sum_{t in chunk} w * enc[b,t,:]
__global__ void __launch_bounds__(256) k_ctx(
    const float* __restrict__ enc, const float* __restrict__ a_out,
    const int* __restrict__ len, float* __restrict__ cpart) {
  const int b = blockIdx.x, ch = blockIdx.y, tid = threadIdx.x;
  const int t0 = ch * 250;
  __shared__ float wsh[256];
  if (tid < 250) {
    int t = t0 + tid;
    float av = a_out[(size_t)b * T_ + t];
    wsh[tid] = (t < len[b]) ? av : 0.f;
  }
  __syncthreads();
  const float4* ep = (const float4*)(enc + (size_t)(b * T_ + t0) * E_);
  float4 acc = {0.f, 0.f, 0.f, 0.f};
  for (int i = 0; i < 250; ++i) {
    float wv = wsh[i];
    float4 x = ep[(size_t)i * 256 + tid];
    acc.x += wv * x.x; acc.y += wv * x.y; acc.z += wv * x.z; acc.w += wv * x.w;
  }
  ((float4*)cpart)[((size_t)(b * 8 + ch)) * 256 + tid] = acc;
}

// ---- 7) reduce the 8 T-chunk partials (deterministic, no atomics) --------
__global__ void __launch_bounds__(256) k_ctx_red(
    const float* __restrict__ cpart, float* __restrict__ c_out) {
  int idx = blockIdx.x * 256 + threadIdx.x;         // 0..32767
  int b = idx >> 10, e = idx & 1023;
  float s = 0.f;
#pragma unroll
  for (int ch = 0; ch < 8; ++ch) s += cpart[((size_t)(b * 8 + ch)) * 1024 + e];
  c_out[idx] = s;
}

// ---- launcher ------------------------------------------------------------
extern "C" void kernel_launch(void* const* d_in, const int* in_sizes, int n_in,
                              void* d_out, int out_size, void* d_ws, size_t ws_size,
                              hipStream_t stream) {
  const float* enc         = (const float*)d_in[0];
  const int*   len         = (const int*)d_in[1];
  const float* dec_z       = (const float*)d_in[2];
  /* d_in[3] att_prev: unused by reference */
  const float* noise       = (const float*)d_in[4];
  const float* W_enc       = (const float*)d_in[5];
  const float* b_enc       = (const float*)d_in[6];
  const float* W_dec       = (const float*)d_in[7];
  const float* g_vec       = (const float*)d_in[8];
  const float* mono_factor = (const float*)d_in[9];
  const float* mono_bias   = (const float*)d_in[10];

  char* ws = (char*)d_ws;
  unsigned short* Wb    = (unsigned short*)ws;                         // 2 MB
  float*          rba   = (float*)(ws + (2u << 20));                   // 128 KB
  float*          e_buf = (float*)(ws + (2u << 20) + (128u << 10));    // 256 KB
  float*          scale = (float*)(ws + (2u << 20) + (384u << 10));    // 1 KB
  float*          cpart = (float*)(ws + (2u << 20) + (385u << 10));    // 1 MB

  float* c_out = (float*)d_out;          // (B,E) = 32768 floats
  float* a_out = c_out + B_ * E_;        // (B,T) = 64000 floats

  k_convert_w<<<1024, 256, 0, stream>>>(W_enc, Wb);
  k_scale<<<1, 256, 0, stream>>>(g_vec, mono_factor, scale);
  k_decproj<<<128, 256, 0, stream>>>(dec_z, W_dec, b_enc, rba);
  k_energy<<<500, 256, 0, stream>>>(enc, Wb, rba, g_vec, scale, mono_bias,
                                    len, noise, e_buf);
  k_scan<<<B_, 256, 0, stream>>>(e_buf, a_out);
  dim3 gctx(B_, 8);
  k_ctx<<<gctx, 256, 0, stream>>>(enc, a_out, len, cpart);
  k_ctx_red<<<128, 256, 0, stream>>>(cpart, c_out);
}
